// MyStackedLSTMWithDropout_85194971283619
// MI455X (gfx1250) — matrix operand order repaired and checked by reading, not compiled, not run on hardware
//
#include <hip/hip_runtime.h>
#include <hip/hip_bf16.h>
#include <math.h>

// ---------------- problem constants ----------------
#define BSZ   64
#define TSZ   1024
#define HSZ   512
#define LSZ   4
#define G4H   2048               // 4*H
#define KD    512                // K depth (=H)
#define NBLK_SCAN 8              // persistent blocks in scan kernel (64 waves)

typedef __attribute__((ext_vector_type(16))) __bf16 v16bf;
typedef __attribute__((ext_vector_type(8)))  __bf16 v8bf;
typedef __attribute__((ext_vector_type(8)))  float  v8f;

// ---------------- WMMA fragment loaders (ISA 7.12.2 layouts) ----------------
// A 16x32 bf16: lane (m + 16*half): V0-3 <- K[kb+8*half .. +7], V4-7 <- K[kb+16+8*half .. +7]
__device__ __forceinline__ v16bf load_afrag(const __bf16* __restrict__ rowptr, int kb, int half) {
    const v8bf lo = *(const v8bf*)(rowptr + kb + 8 * half);
    const v8bf hi = *(const v8bf*)(rowptr + kb + 16 + 8 * half);
    v16bf r;
#pragma unroll
    for (int i = 0; i < 8; ++i) { r[i] = lo[i]; r[i + 8] = hi[i]; }
    return r;
}
// B 32x16 bf16 with B[k][n] = W[n][k] (W row-major, ld=KD):
// lane (n + 16*half): V0-3 <- W[n][kb+8*half .. +7], V4-7 <- W[n][kb+16+8*half .. +7]
__device__ __forceinline__ v16bf load_bfrag(const __bf16* __restrict__ wrowptr, int kb, int half) {
    const v8bf lo = *(const v8bf*)(wrowptr + kb + 8 * half);
    const v8bf hi = *(const v8bf*)(wrowptr + kb + 16 + 8 * half);
    v16bf r;
#pragma unroll
    for (int i = 0; i < 8; ++i) { r[i] = lo[i]; r[i + 8] = hi[i]; }
    return r;
}

__device__ __forceinline__ v8f wmma_bf16(v16bf a, v16bf b, v8f c) {
    return __builtin_amdgcn_wmma_f32_16x16x32_bf16(false, a, false, b, (short)0, c, false, false);
}

__device__ __forceinline__ float sigm(float x) { return 1.0f / (1.0f + expf(-x)); }

// ---------------- conversion kernels ----------------
// inp [B,T,H] f32  ->  Xb [T,B,H] bf16 (time-major swap)
__global__ void k_xpose_cvt(const float* __restrict__ inp, __bf16* __restrict__ xb) {
    int idx = blockIdx.x * 256 + threadIdx.x;           // < T*B*H = 33.5M
    int t = idx / (BSZ * HSZ);
    int r = idx - t * (BSZ * HSZ);
    int b = r / HSZ;
    int u = r - b * HSZ;
    xb[idx] = (__bf16)inp[((size_t)b * TSZ + t) * HSZ + u];
}

__global__ void k_cvt(const float* __restrict__ in, __bf16* __restrict__ out, int n) {
    int i = blockIdx.x * 256 + threadIdx.x;
    if (i < n) out[i] = (__bf16)in[i];
}

// ---------------- input-projection GEMM: xg = X @ Wih^T + (b_ih + b_hh) ----------------
// X:[65536, 512] bf16 row-major, W:[2048, 512] bf16 row-major, xg:[65536, 2048] f32
// 64x32 tile per wave (4 M-frags x 2 N-frags): 8 accum frags (64 VGPRs) + 6 in-flight
// fragments -> spill-free, 0.75 fragment-loads per WMMA.
__global__ void k_gemm_xg(const __bf16* __restrict__ X, const __bf16* __restrict__ W,
                          const float* __restrict__ bih, const float* __restrict__ bhh,
                          float* __restrict__ xg) {
    const int lane = threadIdx.x & 31;
    const int wave = threadIdx.x >> 5;
    const int lr   = lane & 15;
    const int half = lane >> 4;

    const int wid = blockIdx.x * 8 + wave;   // 65536 tiles (1024 M x 64 N)
    const int wn  = wid & 63;                // N tile (of 64)
    const int wm  = wid >> 6;                // M tile (of 1024)
    const int m0  = wm * 64;
    const int n0  = wn * 32;

    v8f acc[4][2];
#pragma unroll
    for (int i = 0; i < 4; ++i)
#pragma unroll
        for (int j = 0; j < 2; ++j) acc[i][j] = (v8f){};

    const __bf16* ap[4];
    const __bf16* bp[2];
#pragma unroll
    for (int i = 0; i < 4; ++i) ap[i] = X + (size_t)(m0 + 16 * i + lr) * KD;
#pragma unroll
    for (int j = 0; j < 2; ++j) bp[j] = W + (size_t)(n0 + 16 * j + lr) * KD;

    for (int kb = 0; kb < KD; kb += 32) {
        v16bf a[4], b[2];
#pragma unroll
        for (int i = 0; i < 4; ++i) a[i] = load_afrag(ap[i], kb, half);
#pragma unroll
        for (int j = 0; j < 2; ++j) b[j] = load_bfrag(bp[j], kb, half);
#pragma unroll
        for (int i = 0; i < 4; ++i)
#pragma unroll
            for (int j = 0; j < 2; ++j) acc[i][j] = wmma_bf16(a[i], b[j], acc[i][j]);
    }

    float bias[2];
#pragma unroll
    for (int j = 0; j < 2; ++j)
        bias[j] = bih[n0 + 16 * j + lr] + bhh[n0 + 16 * j + lr];

#pragma unroll
    for (int i = 0; i < 4; ++i)
#pragma unroll
        for (int r = 0; r < 8; ++r) {
            const int m = m0 + 16 * i + r + 8 * half;
#pragma unroll
            for (int j = 0; j < 2; ++j)
                xg[(size_t)m * G4H + n0 + 16 * j + lr] = acc[i][j][r] + bias[j];
        }
}

// ---------------- per-layer state init: zero h ping buffer + barrier counter ----------------
__global__ void k_init_state(__bf16* __restrict__ hb0, int* __restrict__ cnt) {
    int i = blockIdx.x * 256 + threadIdx.x;
    if (i < BSZ * HSZ) hb0[i] = (__bf16)0.0f;
    if (i == 0) *cnt = 0;
}

// ---------------- software grid barrier ----------------
__device__ __forceinline__ void gbar(int* cnt, int& epoch) {
    __threadfence();                 // release my stores (all threads)
    __syncthreads();
    epoch += NBLK_SCAN;
    if (threadIdx.x == 0) {
        atomicAdd(cnt, 1);
        while (__hip_atomic_load(cnt, __ATOMIC_ACQUIRE, __HIP_MEMORY_SCOPE_AGENT) < epoch) {
            __builtin_amdgcn_s_sleep(2);
        }
        __threadfence();             // acquire others' stores
    }
    __syncthreads();
}

// ---------------- persistent LSTM scan ----------------
// 8 blocks x 8 waves = 64 waves. Wave (wm, grp) owns batch rows [32*wm, 32*wm+32) and
// unit columns [16*grp, 16*grp+16) across ALL FOUR gates (N tiles at g*512 + u0).
// i/f/g/o land in identical lane/VGPR slots of 4 accumulators -> in-register LSTM cell.
// Cell state c lives in registers for all 1024 steps. h double-buffered in bf16.
// One grid barrier per timestep.
__global__ void k_scan(const float* __restrict__ xg, const __bf16* __restrict__ Whh,
                       __bf16* __restrict__ hbA, __bf16* __restrict__ hbB,
                       __bf16* __restrict__ xnext, float* __restrict__ outx,
                       float* __restrict__ sh, float* __restrict__ sc,
                       int* __restrict__ cnt) {
    const int tid  = threadIdx.x;
    const int lane = tid & 31;
    const int wave = tid >> 5;
    const int lr   = lane & 15;
    const int half = lane >> 4;

    const int wid = blockIdx.x * 8 + wave;   // 0..63
    const int grp = wid & 31;                // unit group (of 32)
    const int wm  = wid >> 5;                // 0..1  (batch half)
    const int m0  = wm * 32;
    const int u0  = grp * 16;

    const __bf16* bp[4];                     // Whh rows for gate g, units u0..u0+15
#pragma unroll
    for (int g = 0; g < 4; ++g) bp[g] = Whh + (size_t)(g * HSZ + u0 + lr) * KD;

    float creg[2][8];                        // cell state, resident in VGPRs
#pragma unroll
    for (int mt = 0; mt < 2; ++mt)
#pragma unroll
        for (int r = 0; r < 8; ++r) creg[mt][r] = 0.0f;

    const int u = u0 + lr;
    int epoch = 0;

    for (int t = 0; t < TSZ; ++t) {
        const __bf16* hin  = (t & 1) ? hbB : hbA;
        __bf16*       hout = (t & 1) ? hbA : hbB;
        const float*  xgt  = xg + (size_t)t * BSZ * G4H;

        // --- C init from precomputed input projection ---------------------
        v8f acc[2][4];
#pragma unroll
        for (int mt = 0; mt < 2; ++mt)
#pragma unroll
            for (int r = 0; r < 8; ++r) {
                const int m = m0 + 16 * mt + r + 8 * half;
#pragma unroll
                for (int g = 0; g < 4; ++g)
                    acc[mt][g][r] = xgt[(size_t)m * G4H + g * HSZ + u];
            }

        // --- gate GEMM: acc += h @ Whh^T ----------------------------------
        const __bf16* a0p = hin + (size_t)(m0 + lr) * KD;
        const __bf16* a1p = hin + (size_t)(m0 + 16 + lr) * KD;
        for (int kb = 0; kb < KD; kb += 32) {
            v16bf a0 = load_afrag(a0p, kb, half);
            v16bf a1 = load_afrag(a1p, kb, half);
            v16bf b[4];
#pragma unroll
            for (int g = 0; g < 4; ++g) b[g] = load_bfrag(bp[g], kb, half);
#pragma unroll
            for (int g = 0; g < 4; ++g) {
                acc[0][g] = wmma_bf16(a0, b[g], acc[0][g]);
                acc[1][g] = wmma_bf16(a1, b[g], acc[1][g]);
            }
        }

        // --- in-register LSTM cell (gate order: i, f, g, o) ---------------
#pragma unroll
        for (int mt = 0; mt < 2; ++mt)
#pragma unroll
            for (int r = 0; r < 8; ++r) {
                const float iv = acc[mt][0][r];
                const float fv = acc[mt][1][r];
                const float gv = acc[mt][2][r];
                const float ov = acc[mt][3][r];
                const float cn = sigm(fv) * creg[mt][r] + sigm(iv) * tanhf(gv);
                const float h  = sigm(ov) * tanhf(cn);
                creg[mt][r] = cn;

                const int m = m0 + 16 * mt + r + 8 * half;
                hout[(size_t)m * HSZ + u] = (__bf16)h;
                xnext[(size_t)(t * BSZ + m) * HSZ + u] = (__bf16)h;
                if (outx) outx[(size_t)(t * BSZ + m) * HSZ + u] = h;
                if (t == TSZ - 1) {
                    sh[(size_t)m * HSZ + u] = h;     // final hT (f32)
                    sc[(size_t)m * HSZ + u] = cn;    // final cT (f32)
                }
            }

        gbar(cnt, epoch);   // h fully published before anyone starts step t+1
    }
}

// ---------------- host launcher ----------------
extern "C" void kernel_launch(void* const* d_in, const int* in_sizes, int n_in,
                              void* d_out, int out_size, void* d_ws, size_t ws_size,
                              hipStream_t stream) {
    const float* inp  = (const float*)d_in[0];   // [B,T,H]
    const float* W_ih = (const float*)d_in[1];   // [L,4H,H]
    const float* W_hh = (const float*)d_in[2];   // [L,4H,H]
    const float* b_ih = (const float*)d_in[3];   // [L,4H]
    const float* b_hh = (const float*)d_in[4];   // [L,4H]
    float* fout = (float*)d_out;                 // [T*B*H] + L*([B*H] hT + [B*H] cT)

    const size_t BH   = (size_t)BSZ * HSZ;               // 32768
    const size_t TBH  = (size_t)TSZ * BSZ * HSZ;         // 33,554,432
    const size_t WEL  = (size_t)LSZ * G4H * HSZ;         // 4,194,304 (per weight array)

    // workspace carve-up
    uint8_t* w = (uint8_t*)d_ws;
    int*    cnt  = (int*)w;        w += 256;
    __bf16* Xa   = (__bf16*)w;     w += TBH * 2;
    __bf16* Xb2  = (__bf16*)w;     w += TBH * 2;
    __bf16* Wihb = (__bf16*)w;     w += WEL * 2;
    __bf16* Whhb = (__bf16*)w;     w += WEL * 2;
    float*  xg   = (float*)w;      w += TBH * 4 * sizeof(float);   // [T*B, 4H]
    __bf16* hb0  = (__bf16*)w;     w += BH * 2;
    __bf16* hb1  = (__bf16*)w;     w += BH * 2;

    // convert input (transpose to time-major) and weights to bf16
    k_xpose_cvt<<<(int)(TBH / 256), 256, 0, stream>>>(inp, Xa);
    k_cvt<<<(int)((WEL + 255) / 256), 256, 0, stream>>>(W_ih, Wihb, (int)WEL);
    k_cvt<<<(int)((WEL + 255) / 256), 256, 0, stream>>>(W_hh, Whhb, (int)WEL);

    __bf16* Xcur = Xa;
    __bf16* Xnxt = Xb2;
    for (int l = 0; l < LSZ; ++l) {
        const __bf16* Wi = Wihb + (size_t)l * G4H * HSZ;
        const __bf16* Wh = Whhb + (size_t)l * G4H * HSZ;

        // xg = Xcur @ Wih^T + (b_ih + b_hh)  -- 65536 wave-tiles (64x32) / 8 per block
        k_gemm_xg<<<8192, 256, 0, stream>>>(Xcur, Wi, b_ih + (size_t)l * G4H,
                                            b_hh + (size_t)l * G4H, xg);
        k_init_state<<<128, 256, 0, stream>>>(hb0, cnt);
        k_scan<<<NBLK_SCAN, 256, 0, stream>>>(xg, Wh, hb0, hb1, Xnxt,
                                              (l == LSZ - 1) ? fout : nullptr,
                                              fout + TBH + (size_t)l * 2 * BH,        // hT
                                              fout + TBH + (size_t)l * 2 * BH + BH,   // cT
                                              cnt);

        __bf16* tmp = Xcur; Xcur = Xnxt; Xnxt = tmp;
    }
}
